// AdvancedTransformerBlock_8727373545675
// MI455X (gfx1250) — matrix operand order repaired and checked by reading, not compiled, not run on hardware
//
#include <hip/hip_runtime.h>
#include <hip/hip_bf16.h>

// ---------------------------------------------------------------------------
// Types for CDNA5 WMMA (wave32): v_wmma_f32_16x16x32_bf16
// ---------------------------------------------------------------------------
typedef __attribute__((ext_vector_type(16))) __bf16 v16bf;
typedef __attribute__((ext_vector_type(8)))  float  v8f;

#define D_MODEL  2048
#define D_FF     8192
#define N_HEADS  16
#define HEAD_DIM 128
#define SEQ      2048
#define BATCH    2
#define MROWS    (BATCH * SEQ)   // 4096 token rows

// ---------------------------------------------------------------------------
// Async global->LDS 16B copy (gfx1250 GLOBAL_LOAD_ASYNC_TO_LDS_B128, ASYNCcnt)
// Builtin signature (from hipcc diagnostics): param0 = addrspace(1) v4i*,
// i.e. (global_src, lds_dst, imm_offset, cpol). Falls back to a plain b128
// copy if the builtin is not declared.
// ---------------------------------------------------------------------------
#if defined(__has_builtin)
#if __has_builtin(__builtin_amdgcn_global_load_async_to_lds_b128)
#define HAVE_ASYNC_LDS 1
#endif
#endif

#ifdef HAVE_ASYNC_LDS
typedef int v4i_vs __attribute__((vector_size(16)));
#define ASYNC_CP16(dst, src)                                              \
  __builtin_amdgcn_global_load_async_to_lds_b128(                         \
      (__attribute__((address_space(1))) v4i_vs*)(src),                   \
      (__attribute__((address_space(3))) v4i_vs*)(dst), 0, 0)
#if __has_builtin(__builtin_amdgcn_s_wait_asynccnt)
#define ASYNC_WAIT() __builtin_amdgcn_s_wait_asynccnt(0)
#else
#define ASYNC_WAIT() asm volatile("s_wait_asynccnt 0" ::: "memory")
#endif
#else
#define ASYNC_CP16(dst, src) (*(uint4*)(dst) = *(const uint4*)(src))
#define ASYNC_WAIT() ((void)0)
#endif

// ---------------------------------------------------------------------------
// WMMA fragment loaders from LDS (contiguous k order, same for A and B; not the ISA table order)
// A (16x32 bf16): row = lane&15; 16 contiguous K at (lane>>4)*16
// B (32x16 bf16) stored transposed [col][k]: col = lane&15; 16 contiguous K at (lane>>4)*16
// ---------------------------------------------------------------------------
__device__ __forceinline__ v16bf lds_a_frag(const __bf16* base, int row0, int ld) {
  const int lane = threadIdx.x & 31;
  const int r  = row0 + (lane & 15);
  const int kb = (lane >> 4) * 16;
  union { v16bf v; uint4 q[2]; } u;
  u.q[0] = *(const uint4*)(base + (size_t)r * ld + kb);
  u.q[1] = *(const uint4*)(base + (size_t)r * ld + kb + 8);
  return u.v;
}

__device__ __forceinline__ v16bf lds_b_frag(const __bf16* base, int col0, int ld) {
  const int lane = threadIdx.x & 31;
  const int c  = col0 + (lane & 15);
  const int kb = (lane >> 4) * 16;
  union { v16bf v; uint4 q[2]; } u;
  u.q[0] = *(const uint4*)(base + (size_t)c * ld + kb);
  u.q[1] = *(const uint4*)(base + (size_t)c * ld + kb + 8);
  return u.v;
}

__device__ __forceinline__ v8f wmma_bf16(v16bf a, v16bf b, v8f c) {
  return __builtin_amdgcn_wmma_f32_16x16x32_bf16(false, a, false, b, (short)0, c,
                                                 false, false);
}

__device__ __forceinline__ float gelu_exact(float x) {
  return 0.5f * x * (1.0f + erff(x * 0.70710678118654752f));
}

// ---------------------------------------------------------------------------
// LayerNorm: fp32 in -> bf16 out, one block (256 thr) per token row
// ---------------------------------------------------------------------------
__global__ __launch_bounds__(256)
void ln_kernel(const float* __restrict__ x, const float* __restrict__ g,
               const float* __restrict__ be, __bf16* __restrict__ out, int D) {
  __shared__ float red[256];
  const int row = blockIdx.x;
  const int tid = threadIdx.x;
  const float* xr = x + (size_t)row * D;

  float s = 0.f;
  for (int i = tid; i < D; i += 256) s += xr[i];
  red[tid] = s; __syncthreads();
  for (int st = 128; st > 0; st >>= 1) {
    if (tid < st) red[tid] += red[tid + st];
    __syncthreads();
  }
  const float mu = red[0] / (float)D;
  __syncthreads();

  float vs = 0.f;
  for (int i = tid; i < D; i += 256) { float d = xr[i] - mu; vs += d * d; }
  red[tid] = vs; __syncthreads();
  for (int st = 128; st > 0; st >>= 1) {
    if (tid < st) red[tid] += red[tid + st];
    __syncthreads();
  }
  const float rstd = rsqrtf(red[0] / (float)D + 1e-5f);

  __bf16* orow = out + (size_t)row * D;
  for (int i = tid; i < D; i += 256)
    orow[i] = (__bf16)((xr[i] - mu) * rstd * g[i] + be[i]);
}

// ---------------------------------------------------------------------------
// One-time weight transpose+convert: W f32 [K][N] -> Wt bf16 [N][K]
// 64x64 tiles through LDS so both global read and write are coalesced.
// ---------------------------------------------------------------------------
__global__ __launch_bounds__(256)
void wtrans_kernel(const float* __restrict__ W, __bf16* __restrict__ Wt,
                   int K, int N) {
  __shared__ __bf16 tile[64][72];                 // [n][k], padded
  const int k0 = blockIdx.y * 64, n0 = blockIdx.x * 64;
  const int tid = threadIdx.x;
  {
    const int kk = tid >> 2, nb = (tid & 3) * 16;
    const float* src = W + (size_t)(k0 + kk) * N + n0 + nb;
#pragma unroll
    for (int j = 0; j < 16; ++j) tile[nb + j][kk] = (__bf16)src[j];
  }
  __syncthreads();
  {
    const int nn = tid >> 2, kb = (tid & 3) * 16;
    __bf16* dst = Wt + (size_t)(n0 + nn) * K + k0 + kb;
#pragma unroll
    for (int j = 0; j < 16; ++j) dst[j] = tile[nn][kb + j];
  }
}

// ---------------------------------------------------------------------------
// WMMA GEMM: C[M,N] = act(A_bf16[M,K] x W[K,N] + bias) (+resid_f32)
// 128x128 tile, BK=64, 256 threads = 8 waves; wave -> 32x64 patch (2x4 tiles),
// 16 WMMAs per stage. WTBF: weights pre-transposed bf16 [N][K] -> pure b128
// (async) staging. !WTBF fallback: f32 [K][N] converted during staging.
// ---------------------------------------------------------------------------
template <bool BIAS, bool GELU, bool RESID, bool OUTBF, bool WTBF>
__global__ __launch_bounds__(256)
void gemm_kernel(const __bf16* __restrict__ A, const void* __restrict__ Wv,
                 const float* __restrict__ bias, const float* __restrict__ resid,
                 void* __restrict__ outp, int M, int N, int K) {
  __shared__ __align__(16) __bf16 As[128 * 64];   // [m][k]  16 KB
  __shared__ __align__(16) __bf16 Bs[128 * 64];   // [n][k]  16 KB

  const int tid  = threadIdx.x;
  const int lane = tid & 31;
  const int wid  = tid >> 5;
  const int bm = blockIdx.y * 128;
  const int bn = blockIdx.x * 128;
  const int wr = (wid >> 1) * 32;
  const int wc = (wid & 1) * 64;

  v8f acc[2][4];
  const v8f vzero = {0.f,0.f,0.f,0.f,0.f,0.f,0.f,0.f};
#pragma unroll
  for (int i = 0; i < 2; ++i)
#pragma unroll
    for (int j = 0; j < 4; ++j) acc[i][j] = vzero;

  const int ar = tid >> 1, acb = (tid & 1) * 32;  // A/B b128 staging coords

  for (int k0 = 0; k0 < K; k0 += 64) {
    // ---- Stage A tile 128x64 bf16: 4 async b128 per thread ----
    {
      const __bf16* src = A + (size_t)(bm + ar) * K + k0 + acb;
      __bf16* dst = As + ar * 64 + acb;
#pragma unroll
      for (int j = 0; j < 4; ++j) ASYNC_CP16(dst + j * 8, src + j * 8);
      if (k0 + 64 < K) __builtin_prefetch(src + 64, 0, 1);
    }
    // ---- Stage B tile into Bs[n][k] ----
    if constexpr (WTBF) {
      const __bf16* Wt = (const __bf16*)Wv;       // [N][K] bf16
      const __bf16* src = Wt + (size_t)(bn + ar) * K + k0 + acb;
      __bf16* dst = Bs + ar * 64 + acb;
#pragma unroll
      for (int j = 0; j < 4; ++j) ASYNC_CP16(dst + j * 8, src + j * 8);
      if (k0 + 64 < K) __builtin_prefetch(src + 64, 0, 1);
    } else {
      const float* W = (const float*)Wv;          // [K][N] f32
      const int kk = tid >> 2, nb = (tid & 3) * 32;
      const float* src = W + (size_t)(k0 + kk) * N + bn + nb;
#pragma unroll
      for (int j = 0; j < 32; ++j) Bs[(nb + j) * 64 + kk] = (__bf16)src[j];
    }
    ASYNC_WAIT();
    __syncthreads();

    // ---- 16 WMMAs: 2 row tiles x 4 col tiles x 2 K-steps ----
#pragma unroll
    for (int ks = 0; ks < 2; ++ks) {
      v16bf a0 = lds_a_frag(As + ks * 32, wr,      64);
      v16bf a1 = lds_a_frag(As + ks * 32, wr + 16, 64);
#pragma unroll
      for (int j = 0; j < 4; ++j) {
        v16bf bfr = lds_b_frag(Bs + ks * 32, wc + j * 16, 64);
        acc[0][j] = wmma_bf16(a0, bfr, acc[0][j]);
        acc[1][j] = wmma_bf16(a1, bfr, acc[1][j]);
      }
    }
    __syncthreads();
  }

  // ---- Epilogue: row = v + 8*(lane>>4), col = lane&15 per 16x16 tile ----
  const int rs = (lane >> 4) * 8;
  const int cl = lane & 15;
#pragma unroll
  for (int i = 0; i < 2; ++i) {
#pragma unroll
    for (int j = 0; j < 4; ++j) {
      const int col = bn + wc + j * 16 + cl;
      float bv = 0.f;
      if (BIAS) bv = bias[col];
#pragma unroll
      for (int v = 0; v < 8; ++v) {
        const int row = bm + wr + i * 16 + rs + v;
        float xv = acc[i][j][v] + bv;
        if (GELU)  xv = gelu_exact(xv);
        if (RESID) xv += resid[(size_t)row * N + col];
        if (OUTBF) ((__bf16*)outp)[(size_t)row * N + col] = (__bf16)xv;
        else       ((float*)outp)[(size_t)row * N + col]  = xv;
      }
    }
  }
}

// ---------------------------------------------------------------------------
// Flash attention (bf16 WMMA, f32 online softmax).
// Block = (b, head, 32-query tile); 256 threads = 8 waves.
// Streams 64-key chunks: S = Q·K^T (WMMA) -> LDS, online max/sum per row,
// O += P·V (WMMA) with per-chunk rescale; S never touches HBM.
// ---------------------------------------------------------------------------
__global__ __launch_bounds__(256)
void attn_kernel(const __bf16* __restrict__ q, const __bf16* __restrict__ k,
                 const __bf16* __restrict__ v, __bf16* __restrict__ o) {
  constexpr int QT = 32, KC = 64, HD = HEAD_DIM, D = D_MODEL, S = SEQ;
  __shared__ __align__(16) __bf16 Qs[QT * HD];   //  8 KB  [q][hd]
  __shared__ __align__(16) __bf16 Ks[KC * HD];   // 16 KB  [key][hd]
  __shared__ __align__(16) __bf16 Vs[HD * KC];   // 16 KB  [hd][key]
  __shared__ __align__(16) float  Sb[QT * KC];   //  8 KB
  __shared__ __align__(16) __bf16 Pb[QT * KC];   //  4 KB
  __shared__ float mrow[QT], lrow[QT], crow[QT];

  const int bid = blockIdx.x;
  const int qt = bid & 63;
  const int h  = (bid >> 6) & (N_HEADS - 1);
  const int b  = bid >> 10;
  const int qbase = qt * QT;
  const size_t rowbase = (size_t)b * S;

  const int tid  = threadIdx.x;
  const int lane = tid & 31;
  const int wid  = tid >> 5;
  const int qr0 = (wid >> 2) * 16;
  const int sc0 = (wid & 3) * 16;
  const int oc0 = (wid & 3) * 32;
  const int rs  = (lane >> 4) * 8;
  const int cl  = lane & 15;

  // Load Q tile (32x128) via async LDS copies
  {
    const int qq = tid >> 3, cb = (tid & 7) * 16;
    const __bf16* src = q + (rowbase + qbase + qq) * D + h * HD + cb;
    __bf16* dst = Qs + qq * HD + cb;
    ASYNC_CP16(dst,     src);
    ASYNC_CP16(dst + 8, src + 8);
  }
  if (tid < QT) { mrow[tid] = -1e30f; lrow[tid] = 0.f; }

  const v8f vzero = {0.f,0.f,0.f,0.f,0.f,0.f,0.f,0.f};
  v8f oacc[2] = {vzero, vzero};
  ASYNC_WAIT();
  __syncthreads();

  for (int kc = 0; kc < S; kc += KC) {
    // Stage K chunk [key][hd] (async b128) and V chunk transposed [hd][key]
    {
      const int kk = tid >> 2, cb = (tid & 3) * 32;
      const __bf16* ksrc = k + (rowbase + kc + kk) * D + h * HD + cb;
      __bf16* kdst = Ks + kk * HD + cb;
#pragma unroll
      for (int j = 0; j < 4; ++j) ASYNC_CP16(kdst + j * 8, ksrc + j * 8);
      const __bf16* vsrc = v + (rowbase + kc + kk) * D + h * HD + cb;
#pragma unroll
      for (int j = 0; j < 32; ++j) Vs[(cb + j) * KC + kk] = vsrc[j];
    }
    ASYNC_WAIT();
    __syncthreads();

    // S = Q · K^T (each wave one 16x16 tile, 4 k-steps over hd=128)
    v8f sacc = vzero;
#pragma unroll
    for (int ks = 0; ks < 4; ++ks) {
      v16bf a = lds_a_frag(Qs + ks * 32, qr0, HD);
      v16bf bfr = lds_b_frag(Ks + ks * 32, sc0, HD);
      sacc = wmma_bf16(a, bfr, sacc);
    }
#pragma unroll
    for (int vv = 0; vv < 8; ++vv)
      Sb[(qr0 + rs + vv) * KC + sc0 + cl] = sacc[vv] * 0.0883883476483184f; // 1/sqrt(128)
    __syncthreads();

    // Online softmax row update (one thread per query row)
    if (tid < QT) {
      float m = mrow[tid];
      float mx = m;
      const float* srow = Sb + tid * KC;
      for (int j = 0; j < KC; ++j) mx = fmaxf(mx, srow[j]);
      const float c = __expf(m - mx);
      float sum = 0.f;
      __bf16* prow = Pb + tid * KC;
      for (int j = 0; j < KC; ++j) {
        const float p = __expf(srow[j] - mx);
        prow[j] = (__bf16)p;
        sum += p;
      }
      lrow[tid] = lrow[tid] * c + sum;
      mrow[tid] = mx;
      crow[tid] = c;
    }
    __syncthreads();

    // O = O*c + P · V (each wave: 2 O tiles, 2 k-steps over KC=64)
#pragma unroll
    for (int t = 0; t < 2; ++t)
#pragma unroll
      for (int vv = 0; vv < 8; ++vv)
        oacc[t][vv] *= crow[qr0 + rs + vv];
#pragma unroll
    for (int ks = 0; ks < 2; ++ks) {
      v16bf a = lds_a_frag(Pb + ks * 32, qr0, KC);
#pragma unroll
      for (int t = 0; t < 2; ++t) {
        v16bf bfr = lds_b_frag(Vs + ks * 32, oc0 + t * 16, KC);
        oacc[t] = wmma_bf16(a, bfr, oacc[t]);
      }
    }
    __syncthreads();
  }

  // Normalize and store O (bf16, [B*S, D] with head offset)
#pragma unroll
  for (int t = 0; t < 2; ++t) {
#pragma unroll
    for (int vv = 0; vv < 8; ++vv) {
      const int r = qr0 + rs + vv;
      const float val = oacc[t][vv] / lrow[r];
      o[(rowbase + qbase + r) * D + h * HD + oc0 + t * 16 + cl] = (__bf16)val;
    }
  }
}

// ---------------------------------------------------------------------------
// Host: full pre-LN transformer block
// ---------------------------------------------------------------------------
extern "C" void kernel_launch(void* const* d_in, const int* in_sizes, int n_in,
                              void* d_out, int out_size, void* d_ws, size_t ws_size,
                              hipStream_t stream) {
  const float* x   = (const float*)d_in[0];
  const float* wq  = (const float*)d_in[1];
  const float* wk  = (const float*)d_in[2];
  const float* wv  = (const float*)d_in[3];
  const float* wo  = (const float*)d_in[4];
  const float* w1  = (const float*)d_in[5];
  const float* b1  = (const float*)d_in[6];
  const float* w2  = (const float*)d_in[7];
  const float* b2  = (const float*)d_in[8];
  const float* g1  = (const float*)d_in[9];
  const float* be1 = (const float*)d_in[10];
  const float* g2  = (const float*)d_in[11];
  const float* be2 = (const float*)d_in[12];

  const size_t tokD  = (size_t)MROWS * D_MODEL;
  const size_t tokFF = (size_t)MROWS * D_FF;
  const size_t wD    = (size_t)D_MODEL * D_MODEL;   // 4.2M elems
  const size_t wFFsz = (size_t)D_MODEL * D_FF;      // 16.8M elems

  const size_t need_base = tokD * 2 * 6 + tokD * 4 + tokFF * 2;      // ~201 MB
  const size_t need_w    = (wD * 4 + wFFsz * 2) * sizeof(__bf16);    // ~101 MB
  const bool pret = ws_size >= need_base + need_w;

  char* ws = (char*)d_ws;
  __bf16* h1   = (__bf16*)ws; ws += tokD * sizeof(__bf16);
  __bf16* qb   = (__bf16*)ws; ws += tokD * sizeof(__bf16);
  __bf16* kb   = (__bf16*)ws; ws += tokD * sizeof(__bf16);
  __bf16* vb   = (__bf16*)ws; ws += tokD * sizeof(__bf16);
  __bf16* attb = (__bf16*)ws; ws += tokD * sizeof(__bf16);
  float*  x1   = (float*)ws;  ws += tokD * sizeof(float);
  __bf16* h2   = (__bf16*)ws; ws += tokD * sizeof(__bf16);
  __bf16* ff1  = (__bf16*)ws; ws += tokFF * sizeof(__bf16);
  __bf16* wqt  = (__bf16*)ws; ws += wD * sizeof(__bf16);
  __bf16* wkt  = (__bf16*)ws; ws += wD * sizeof(__bf16);
  __bf16* wvt  = (__bf16*)ws; ws += wD * sizeof(__bf16);
  __bf16* wot  = (__bf16*)ws; ws += wD * sizeof(__bf16);
  __bf16* w1t  = (__bf16*)ws; ws += wFFsz * sizeof(__bf16);
  __bf16* w2t  = (__bf16*)ws; ws += wFFsz * sizeof(__bf16);

  const dim3 blk(256);
  const dim3 gD(D_MODEL / 128, MROWS / 128);   // (16, 32)
  const dim3 gFF(D_FF / 128,   MROWS / 128);   // (64, 32)

  // 0) One-time weight transpose+convert (f32 [K][N] -> bf16 [N][K])
  if (pret) {
    const dim3 gW (D_MODEL / 64, D_MODEL / 64);  // (32, 32)
    const dim3 gW1(D_FF / 64,    D_MODEL / 64);  // (128, 32)  w1: K=D, N=FF
    const dim3 gW2(D_MODEL / 64, D_FF / 64);     // (32, 128)  w2: K=FF, N=D
    wtrans_kernel<<<gW,  blk, 0, stream>>>(wq, wqt, D_MODEL, D_MODEL);
    wtrans_kernel<<<gW,  blk, 0, stream>>>(wk, wkt, D_MODEL, D_MODEL);
    wtrans_kernel<<<gW,  blk, 0, stream>>>(wv, wvt, D_MODEL, D_MODEL);
    wtrans_kernel<<<gW,  blk, 0, stream>>>(wo, wot, D_MODEL, D_MODEL);
    wtrans_kernel<<<gW1, blk, 0, stream>>>(w1, w1t, D_MODEL, D_FF);
    wtrans_kernel<<<gW2, blk, 0, stream>>>(w2, w2t, D_FF, D_MODEL);
  }

  // 1) LN1
  ln_kernel<<<MROWS, blk, 0, stream>>>(x, g1, be1, h1, D_MODEL);

  // 2) Q, K, V projections (bf16 out)
  if (pret) {
    gemm_kernel<false,false,false,true,true><<<gD, blk, 0, stream>>>(
        h1, wqt, nullptr, nullptr, qb, MROWS, D_MODEL, D_MODEL);
    gemm_kernel<false,false,false,true,true><<<gD, blk, 0, stream>>>(
        h1, wkt, nullptr, nullptr, kb, MROWS, D_MODEL, D_MODEL);
    gemm_kernel<false,false,false,true,true><<<gD, blk, 0, stream>>>(
        h1, wvt, nullptr, nullptr, vb, MROWS, D_MODEL, D_MODEL);
  } else {
    gemm_kernel<false,false,false,true,false><<<gD, blk, 0, stream>>>(
        h1, wq, nullptr, nullptr, qb, MROWS, D_MODEL, D_MODEL);
    gemm_kernel<false,false,false,true,false><<<gD, blk, 0, stream>>>(
        h1, wk, nullptr, nullptr, kb, MROWS, D_MODEL, D_MODEL);
    gemm_kernel<false,false,false,true,false><<<gD, blk, 0, stream>>>(
        h1, wv, nullptr, nullptr, vb, MROWS, D_MODEL, D_MODEL);
  }

  // 3) Flash attention: B * H * (S/32) = 2048 blocks
  attn_kernel<<<BATCH * N_HEADS * (SEQ / 32), blk, 0, stream>>>(qb, kb, vb, attb);

  // 4) Output projection + residual (f32 out)
  if (pret)
    gemm_kernel<false,false,true,false,true><<<gD, blk, 0, stream>>>(
        attb, wot, nullptr, x, x1, MROWS, D_MODEL, D_MODEL);
  else
    gemm_kernel<false,false,true,false,false><<<gD, blk, 0, stream>>>(
        attb, wo, nullptr, x, x1, MROWS, D_MODEL, D_MODEL);

  // 5) LN2
  ln_kernel<<<MROWS, blk, 0, stream>>>(x1, g2, be2, h2, D_MODEL);

  // 6) FF1: GELU(h2 @ W1 + b1) -> bf16
  if (pret)
    gemm_kernel<true,true,false,true,true><<<gFF, blk, 0, stream>>>(
        h2, w1t, b1, nullptr, ff1, MROWS, D_FF, D_MODEL);
  else
    gemm_kernel<true,true,false,true,false><<<gFF, blk, 0, stream>>>(
        h2, w1, b1, nullptr, ff1, MROWS, D_FF, D_MODEL);

  // 7) FF2: ff1 @ W2 + b2 + residual -> f32 output
  if (pret)
    gemm_kernel<true,false,true,false,true><<<gD, blk, 0, stream>>>(
        ff1, w2t, b2, x1, (float*)d_out, MROWS, D_MODEL, D_FF);
  else
    gemm_kernel<true,false,true,false,false><<<gD, blk, 0, stream>>>(
        ff1, w2, b2, x1, (float*)d_out, MROWS, D_MODEL, D_FF);
}
